// WeavingLoomArchitecture_42245298323789
// MI455X (gfx1250) — matrix operand order repaired and checked by reading, not compiled, not run on hardware
//
#include <hip/hip_runtime.h>
#include <hip/hip_bf16.h>

typedef __attribute__((ext_vector_type(16))) _Float16 v16h;
typedef __attribute__((ext_vector_type(8)))  _Float16 v8h;
typedef __attribute__((ext_vector_type(8)))  float    v8f;

// ---------------------------------------------------------------- helpers ---

__device__ __forceinline__ v16h cat8(v8h lo, v8h hi) {
  return __builtin_shufflevector(lo, hi, 0,1,2,3,4,5,6,7,8,9,10,11,12,13,14,15);
}

__device__ __forceinline__ v8f wmma_f16(v16h a, v16h b, v8f c) {
  // D = A(16x32 f16) * B(32x16 f16) + C(16x16 f32)
  return __builtin_amdgcn_wmma_f32_16x16x32_f16(
      /*neg_a=*/false, a, /*neg_b=*/false, b,
      /*c_mod=*/(short)0, c, /*reuse_a=*/false, /*reuse_b=*/false);
}

// CDNA5 async global->LDS copy (ASYNCcnt path), portable inline-asm form.
__device__ __forceinline__ void async_ld_b128(const void* g, unsigned lds_off) {
  asm volatile("global_load_async_to_lds_b128 %0, %1, off"
               :: "v"(lds_off), "v"((unsigned long long)(size_t)g)
               : "memory");
}
__device__ __forceinline__ void wait_async0() {
  asm volatile("s_wait_asynccnt 0" ::: "memory");
}
// double-buffer wait: async loads complete in order, so waiting ASYNCcnt<=4
// retires the older stage (4 older loads) while the 4 newer remain in flight.
__device__ __forceinline__ void wait_async_le4() {
  asm volatile("s_wait_asynccnt 4" ::: "memory");
}

// ------------------------------------------------------------ tiny kernels ---

__global__ __launch_bounds__(256)
void f32_to_f16_kernel(const float* __restrict__ src, _Float16* __restrict__ dst,
                       size_t n) {
  size_t i = (size_t)blockIdx.x * blockDim.x + threadIdx.x;
  if (i < n) dst[i] = (_Float16)src[i];
}

// Level 1: gate = sigmoid(x . wg + bg); gx = f16(x * gate). One block per row.
__global__ __launch_bounds__(256)
void gate_cast_kernel(const float* __restrict__ x, const float* __restrict__ wg,
                      const float* __restrict__ bg, _Float16* __restrict__ gx,
                      int D) {
  __shared__ float red[256];
  const size_t base = (size_t)blockIdx.x * D;
  float p = 0.f;
  for (int d = threadIdx.x; d < D; d += 256) p += x[base + d] * wg[d];
  red[threadIdx.x] = p;
  __syncthreads();
  for (int s = 128; s > 0; s >>= 1) {
    if ((int)threadIdx.x < s) red[threadIdx.x] += red[threadIdx.x + s];
    __syncthreads();
  }
  const float gate = 1.f / (1.f + __expf(-(red[0] + bg[0])));
  for (int d = threadIdx.x; d < D; d += 256)
    gx[base + d] = (_Float16)(x[base + d] * gate);
}

// --------------------------------------------------------------- WMMA GEMM ---
// C(MxN) = A(MxK,f16,row-major) * W(NxK,f16,row-major)^T, f32 accumulate.
// Block = 256 threads = 8 waves; tile 128(M) x 128(N); K-step 64.
// W tiles double-buffered in LDS via async-LDS loads (copy overlaps WMMA);
// A read direct from global (L2-resident), prefetched ahead.

#define GEMM_KS 64
#define GEMM_BPAD 8                       // row stride 72 halves = 144B (9x16B)
typedef _Float16 BRow[GEMM_KS + GEMM_BPAD];

__device__ __forceinline__ void stage_btile(const _Float16* __restrict__ W,
                                            int K, int n0, int kk,
                                            BRow* dst, int tid) {
  // 128 rows x 64 halves = 1024 x 16B chunks; 4 chunks per thread
#pragma unroll
  for (int s = 0; s < 4; ++s) {
    const int idx = tid + s * 256;
    const int row = idx >> 3;             // 0..127
    const int ch  = (idx & 7) * 8;        // halves 0,8,...,56
    async_ld_b128(W + (size_t)(n0 + row) * K + kk + ch,
                  (unsigned)(size_t)(&dst[row][ch]));
  }
}

__global__ __launch_bounds__(256)
void gemm_wmma_kernel(const _Float16* __restrict__ A,
                      const _Float16* __restrict__ W,
                      _Float16* __restrict__ outH, float* __restrict__ outF,
                      int M, int N, int K) {
  __shared__ __align__(16) _Float16 ldsB[2][128][GEMM_KS + GEMM_BPAD];  // 36 KB
  (void)M;

  const int lane = threadIdx.x & 31;
  const int wave = threadIdx.x >> 5;
  const int r16  = lane & 15;
  const int hi   = lane >> 4;

  const int n0 = blockIdx.x * 128;
  const int m0 = blockIdx.y * 128 + wave * 16;

  v8f acc[8] = {};

  const _Float16* Arow = A + (size_t)(m0 + r16) * K;

  stage_btile(W, K, n0, 0, ldsB[0], threadIdx.x);

  for (int kk = 0; kk < K; kk += GEMM_KS) {
    const int  cur  = (kk / GEMM_KS) & 1;
    const bool more = (kk + GEMM_KS) < K;
    if (more) {
      stage_btile(W, K, n0, kk + GEMM_KS, ldsB[cur ^ 1], threadIdx.x);
      __builtin_prefetch(Arow + kk + 2 * GEMM_KS, 0, 0);
      wait_async_le4();            // retire current stage, keep next in flight
    } else {
      wait_async0();
    }
    __syncthreads();

#pragma unroll
    for (int kh = 0; kh < GEMM_KS; kh += 32) {
      // A fragment: row = r16, 16 consecutive K at kh+hi*16
      v16h a = cat8(*(const v8h*)(Arow + kk + kh + hi * 16),
                    *(const v8h*)(Arow + kk + kh + hi * 16 + 8));
#pragma unroll
      for (int j = 0; j < 8; ++j) {
        // B fragment: col = r16 (16-wide slice j), 16 consecutive K at kh+hi*16
        const _Float16* bp = &ldsB[cur][j * 16 + r16][kh + hi * 16];
        v16h bb = cat8(*(const v8h*)bp, *(const v8h*)(bp + 8));
        acc[j] = wmma_f16(a, bb, acc[j]);
      }
    }
    __syncthreads();
  }

#pragma unroll
  for (int j = 0; j < 8; ++j) {
    const int col = n0 + j * 16 + r16;
#pragma unroll
    for (int vI = 0; vI < 8; ++vI) {
      const size_t idx = (size_t)(m0 + vI + hi * 8) * N + col;
      if (outH) outH[idx] = (_Float16)acc[j][vI];
      if (outF) outF[idx] = acc[j][vI];
    }
  }
}

// ---------------------------------------------- Level 2: 8-token block attn ---
// One block per 8-token group. Scores + bias + softmax + PV, all f32 VALU.

__global__ __launch_bounds__(256)
void block_attn_kernel(const _Float16* __restrict__ q2,
                       const _Float16* __restrict__ k2,
                       const _Float16* __restrict__ v2,
                       const float* __restrict__ bias2,
                       _Float16* __restrict__ o2, int D, float scale) {
  __shared__ float part[64][4];
  __shared__ float sP[8][8];
  const int t0 = blockIdx.x * 8;
  const int pair = threadIdx.x >> 2;  // (q,k) pair 0..63
  const int qtr  = threadIdx.x & 3;   // quarter of D
  const int qi = pair >> 3, ki = pair & 7;
  const _Float16* qrow = q2 + (size_t)(t0 + qi) * D;
  const _Float16* krow = k2 + (size_t)(t0 + ki) * D;
  float p = 0.f;
  const int dq = D >> 2;
  for (int d = qtr * dq; d < (qtr + 1) * dq; ++d)
    p += (float)qrow[d] * (float)krow[d];
  part[pair][qtr] = p;
  __syncthreads();
  if (threadIdx.x < 64) {
    const int qq = threadIdx.x >> 3, kk = threadIdx.x & 7;
    part[threadIdx.x][0] =
        (part[threadIdx.x][0] + part[threadIdx.x][1] + part[threadIdx.x][2] +
         part[threadIdx.x][3]) * scale + bias2[qq * 8 + kk];
  }
  __syncthreads();
  if (threadIdx.x < 8) {
    const int qq = threadIdx.x;
    float mx = -1e30f;
    for (int k = 0; k < 8; ++k) mx = fmaxf(mx, part[qq * 8 + k][0]);
    float e[8], sum = 0.f;
    for (int k = 0; k < 8; ++k) { e[k] = __expf(part[qq * 8 + k][0] - mx); sum += e[k]; }
    const float inv = 1.f / sum;
    for (int k = 0; k < 8; ++k) sP[qq][k] = e[k] * inv;
  }
  __syncthreads();
  for (int d = threadIdx.x; d < D; d += 256) {
    float vv[8];
#pragma unroll
    for (int k = 0; k < 8; ++k) vv[k] = (float)v2[(size_t)(t0 + k) * D + d];
#pragma unroll
    for (int qq = 0; qq < 8; ++qq) {
      float o = 0.f;
#pragma unroll
      for (int k = 0; k < 8; ++k) o += sP[qq][k] * vv[k];
      o2[(size_t)(t0 + qq) * D + d] = (_Float16)o;
    }
  }
}

// --------------------------------------- Level 3: flash attention via WMMA ---
// Block = 256 thr = 8 waves; 16 query rows / block; 32-key steps.
// Waves split D=1024 for QK^T partials (ds_add_f32 reduction), each wave owns
// a 128-wide slice of O (8 x v8f accumulators) for P*V.

__global__ __launch_bounds__(256)
void flash_attn3_kernel(const _Float16* __restrict__ Q,
                        const _Float16* __restrict__ Km,
                        const _Float16* __restrict__ V,
                        _Float16* __restrict__ O, int T, int D, float scale) {
  __shared__ float sS[16][32];
  __shared__ __align__(16) _Float16 sP[16][40];  // padded row stride (80B)
  __shared__ float mrow[16], lrow[16], alphaS[16];

  const int lane = threadIdx.x & 31;
  const int wave = threadIdx.x >> 5;
  const int r16  = lane & 15;
  const int hi   = lane >> 4;
  const int b    = blockIdx.y;
  const int t0   = blockIdx.x * 16;
  const int dsl  = wave * 128;

  const _Float16* Qb = Q  + (size_t)b * T * D;
  const _Float16* Kb = Km + (size_t)b * T * D;
  const _Float16* Vb = V  + (size_t)b * T * D;
  const _Float16* qrow = Qb + (size_t)(t0 + r16) * D;

  v8f acc[8] = {};

  if (threadIdx.x < 16) { mrow[threadIdx.x] = -1e30f; lrow[threadIdx.x] = 0.f; }
  __syncthreads();

  for (int k0 = 0; k0 < T; k0 += 32) {
    // zero score tile
    float* sflat = &sS[0][0];
    for (int i = threadIdx.x; i < 16 * 32; i += 256) sflat[i] = 0.f;
    __syncthreads();

    // partial S = Q * K^T over this wave's 128-wide d slice
    v8f s0 = {}, s1 = {};
#pragma unroll
    for (int dk = 0; dk < 128; dk += 32) {
      const int d = dsl + dk;
      v16h qa = cat8(*(const v8h*)(qrow + d + hi * 16),
                     *(const v8h*)(qrow + d + hi * 16 + 8));
      const _Float16* kr0 = Kb + (size_t)(k0 + r16) * D + d + hi * 16;
      const _Float16* kr1 = kr0 + (size_t)16 * D;
      v16h kb0 = cat8(*(const v8h*)kr0, *(const v8h*)(kr0 + 8));
      v16h kb1 = cat8(*(const v8h*)kr1, *(const v8h*)(kr1 + 8));
      s0 = wmma_f16(qa, kb0, s0);
      s1 = wmma_f16(qa, kb1, s1);
    }
#pragma unroll
    for (int vI = 0; vI < 8; ++vI) {
      atomicAdd(&sS[vI + hi * 8][r16],      s0[vI]);
      atomicAdd(&sS[vI + hi * 8][16 + r16], s1[vI]);
    }
    __syncthreads();

    // online softmax update for the 16 rows
    if (threadIdx.x < 16) {
      const int qq = threadIdx.x;
      const float mold = mrow[qq];
      float mx = mold;
      for (int kk = 0; kk < 32; ++kk) mx = fmaxf(mx, sS[qq][kk] * scale);
      const float alpha = __expf(mold - mx);
      float lsum = 0.f;
      for (int kk = 0; kk < 32; ++kk) {
        const float e = __expf(sS[qq][kk] * scale - mx);
        sP[qq][kk] = (_Float16)e;
        lsum += e;
      }
      lrow[qq] = lrow[qq] * alpha + lsum;
      mrow[qq] = mx;
      alphaS[qq] = alpha;
    }
    __syncthreads();

    // rescale accumulators
    float al[8];
#pragma unroll
    for (int vI = 0; vI < 8; ++vI) al[vI] = alphaS[vI + hi * 8];
#pragma unroll
    for (int j = 0; j < 8; ++j)
#pragma unroll
      for (int vI = 0; vI < 8; ++vI) acc[j][vI] *= al[vI];

    // O += P * V   (A frag = P from LDS; B frags = V columns, strided)
    const _Float16* pp = &sP[r16][0];
    v16h pa = cat8(*(const v8h*)(pp + hi * 16), *(const v8h*)(pp + hi * 16 + 8));
    const _Float16* vb0 = Vb + (size_t)(k0 + hi * 16) * D + dsl + r16;
#pragma unroll
    for (int j = 0; j < 8; ++j) {
      const _Float16* vp = vb0 + j * 16;
      v16h vbf;
#pragma unroll
      for (int i = 0; i < 16; ++i) vbf[i] = vp[(size_t)i * D];
      acc[j] = wmma_f16(pa, vbf, acc[j]);
    }
  }

  // finalize: divide by l, store o3 (f16)
  float linv[8];
#pragma unroll
  for (int vI = 0; vI < 8; ++vI) linv[vI] = 1.f / lrow[vI + hi * 8];
  _Float16* ob = O + ((size_t)b * T + t0) * D;
#pragma unroll
  for (int j = 0; j < 8; ++j) {
    const int col = dsl + j * 16 + r16;
#pragma unroll
    for (int vI = 0; vI < 8; ++vI)
      ob[(size_t)(vI + hi * 8) * D + col] = (_Float16)(acc[j][vI] * linv[vI]);
  }
}

// ----------------------------------------------------------------- launcher --

extern "C" void kernel_launch(void* const* d_in, const int* in_sizes, int n_in,
                              void* d_out, int out_size, void* d_ws, size_t ws_size,
                              hipStream_t stream) {
  (void)in_sizes; (void)n_in; (void)out_size; (void)ws_size;

  constexpr int Bb = 4, T = 2048, D = 1024;
  constexpr size_t MT = (size_t)Bb * T;          // 8192 rows
  constexpr float scale = 0.03125f;              // D^-0.5

  const float* x  = (const float*)d_in[0];
  const float* wg = (const float*)d_in[1];
  const float* bg = (const float*)d_in[2];
  const float* wsrc[7] = {
      (const float*)d_in[3], (const float*)d_in[4], (const float*)d_in[5],
      (const float*)d_in[6], (const float*)d_in[7], (const float*)d_in[8],
      (const float*)d_in[9]};
  const float* bias2 = (const float*)d_in[10];
  float* out = (float*)d_out;

  char* ws = (char*)d_ws;
  size_t off = 0;
  auto alloc = [&](size_t bytes) {
    void* p = ws + off;
    off = (off + bytes + 255) & ~(size_t)255;
    return p;
  };

  _Float16* w16[7];
  for (int i = 0; i < 7; ++i) w16[i] = (_Float16*)alloc((size_t)D * D * 2);
  _Float16* gx = (_Float16*)alloc(MT * D * 2);  // gated x; later reused as o3
  _Float16* qb = (_Float16*)alloc(MT * D * 2);  // q2 then q3
  _Float16* kb = (_Float16*)alloc(MT * D * 2);  // k2 then k3
  _Float16* vb = (_Float16*)alloc(MT * D * 2);  // v2 then v3
  _Float16* o2 = (_Float16*)alloc(MT * D * 2);

  // 1) weights fp32 -> fp16
  {
    const size_t n = (size_t)D * D;
    const int thr = 256, blk = (int)((n + thr - 1) / thr);
    for (int i = 0; i < 7; ++i)
      f32_to_f16_kernel<<<blk, thr, 0, stream>>>(wsrc[i], w16[i], n);
  }
  // 2) level-1 gate + cast
  gate_cast_kernel<<<(int)MT, 256, 0, stream>>>(x, wg, bg, gx, D);
  // 3) q2/k2/v2 projections
  dim3 ggrid(D / 128, (unsigned)(MT / 128));
  gemm_wmma_kernel<<<ggrid, 256, 0, stream>>>(gx, w16[0], qb, nullptr, (int)MT, D, D);
  gemm_wmma_kernel<<<ggrid, 256, 0, stream>>>(gx, w16[1], kb, nullptr, (int)MT, D, D);
  gemm_wmma_kernel<<<ggrid, 256, 0, stream>>>(gx, w16[2], vb, nullptr, (int)MT, D, D);
  // 4) level-2 local attention
  block_attn_kernel<<<(int)(MT / 8), 256, 0, stream>>>(qb, kb, vb, bias2, o2, D, scale);
  // 5) q3/k3/v3 projections (reuse q/k/v buffers)
  gemm_wmma_kernel<<<ggrid, 256, 0, stream>>>(o2, w16[3], qb, nullptr, (int)MT, D, D);
  gemm_wmma_kernel<<<ggrid, 256, 0, stream>>>(o2, w16[4], kb, nullptr, (int)MT, D, D);
  gemm_wmma_kernel<<<ggrid, 256, 0, stream>>>(o2, w16[5], vb, nullptr, (int)MT, D, D);
  // 6) level-3 flash attention -> o3 (reuse gx)
  dim3 agrid(T / 16, Bb);
  flash_attn3_kernel<<<agrid, 256, 0, stream>>>(qb, kb, vb, gx, T, D, scale);
  // 7) final projection, fp32 output
  gemm_wmma_kernel<<<ggrid, 256, 0, stream>>>(gx, w16[6], nullptr, out, (int)MT, D, D);
}
